// PointNet4D_26087631356327
// MI455X (gfx1250) — matrix operand order repaired and checked by reading, not compiled, not run on hardware
//
#include <hip/hip_runtime.h>
#include <hip/hip_bf16.h>
#include <math.h>

// ---------------- problem dims ----------------
#define LSEQ   150
#define NBATCH 16
#define DMODEL 1024
#define DINNER 2048          // mamba expand*d_model
#define NSTATE 16
#define DTRANK 64
#define MTOK   (NBATCH*LSEQ) // 2400 tokens
#define NHEADS 4
#define DHEAD  1024
#define AINNER 4096          // heads*dim_head
#define MLPH   4096

// ---------------- WMMA types ----------------
typedef __attribute__((ext_vector_type(16))) __bf16        v16bf;
typedef __attribute__((ext_vector_type(8)))  float         v8f;
typedef __attribute__((ext_vector_type(4)))  unsigned int  u32x4;

union Frag16 { v16bf v; u32x4 q[2]; };

// pack two f32 -> packed bf16 dword using CDNA5 VOP3P mixed-precision FMA
// (v_fma_mixlo_bf16 / v_fma_mixhi_bf16: cvt(fma(s0,1.0,0)) with HW RNE)
__device__ __forceinline__ unsigned pack_bf16x2(float lo, float hi) {
  unsigned d;
  asm("v_fma_mixlo_bf16 %0, %1, 1.0, 0" : "=v"(d) : "v"(lo));
  asm("v_fma_mixhi_bf16 %0, %1, 1.0, 0" : "+v"(d) : "v"(hi));
  return d;
}

// load 8 consecutive-k f32 elements of a row-major operand (zero-filled OOB)
__device__ __forceinline__ void load_unit_rowmajor(
    const float* __restrict__ src, int ld, int R, int K,
    int gr, int gk, float* __restrict__ f)
{
  if (gr < R) {
    const float* p = src + (long long)gr * ld + gk;
    if (gk + 7 < K) {
      #pragma unroll
      for (int j = 0; j < 8; j += 2) {               // 8B-aligned b64 loads
        float2 t = *(const float2*)(p + j);
        f[j] = t.x; f[j + 1] = t.y;
      }
    } else {
      #pragma unroll
      for (int j = 0; j < 8; ++j) f[j] = (gk + j < K) ? p[j] : 0.f;
    }
  } else {
    #pragma unroll
    for (int j = 0; j < 8; ++j) f[j] = 0.f;
  }
}

// pack 8 f32 -> 8 bf16 and store as one b128 to LDS
__device__ __forceinline__ void store_unit(unsigned short* __restrict__ p,
                                           const float* __restrict__ f)
{
  u32x4 w;
  #pragma unroll
  for (int j = 0; j < 4; ++j)
    w[j] = pack_bf16x2(f[2 * j], f[2 * j + 1]);
  *(u32x4*)p = w;
}

// =====================================================================
// Generic batched GEMM:  C[M,N] = act(A[M,K] * op(B) + bias) + residual
//   op(B): bTrans==0 -> B[n*ldb+k]  (weights stored [N,K])
//          bTrans==1 -> B[k*ldb+n]  (contraction-major, for attn @ V)
// Batch index z = bb*nbh + bh, with independent (bb,bh) strides per operand.
// bf16 inputs staged in double-buffered LDS (packed b128 stores), f32
// accumulation via v_wmma_f32_16x16x32_bf16.
// Block: 256 thr = 8 waves; tile 128(M) x 64(N); wave tile 32x32 (2x2 WMMA).
// =====================================================================
__global__ __launch_bounds__(256) void wmma_gemm_bf16(
    const float* __restrict__ A, const float* __restrict__ Bm, float* __restrict__ C,
    const float* __restrict__ bias, const float* __restrict__ residual,
    int M, int N, int K, int lda, int ldb, int ldc,
    long long sAb, long long sAh, long long sBb, long long sBh,
    long long sCb, long long sCh, int nbh, int bTrans, int act)
{
  __shared__ unsigned short As[2][128 * 32];   // 2 x 8KB
  __shared__ unsigned short Bs[2][64 * 32];    // 2 x 4KB

  int z  = blockIdx.z;
  int bb = z / nbh, bh = z % nbh;
  const float* Ab = A  + bb * sAb + bh * sAh;
  const float* Bb = Bm + bb * sBb + bh * sBh;
  float*       Cb = C  + bb * sCb + bh * sCh;

  int m0 = blockIdx.y * 128;
  int n0 = blockIdx.x * 64;
  int tid  = threadIdx.x;
  int lane = tid & 31, wave = tid >> 5;
  int wm = wave >> 1, wn = wave & 1;       // 4x2 wave grid
  int hr = lane & 15, hi = lane >> 4;

  // per-thread staging assignments (8-element "units" along k)
  // A: 128 rows x 4 units -> 512 units, 2 per thread
  const int ar0 = tid >> 2,          ac0 = (tid & 3) * 8;
  const int ar1 = (tid + 256) >> 2,  ac1 = ac0;            // same col unit, +64 rows
  // B: 64 rows x 4 units -> 256 units, 1 per thread
  const int brw = tid >> 2,          bcu = (tid & 3) * 8;

  v8f acc[2][2];
  #pragma unroll
  for (int i = 0; i < 2; ++i)
    #pragma unroll
    for (int j = 0; j < 2; ++j)
      #pragma unroll
      for (int e = 0; e < 8; ++e) acc[i][j][e] = 0.f;

  float aReg[2][8], bReg[8];
  const int nk = (K + 31) / 32;

  // ---- fetch chunk 0 into registers, commit to buffer 0 ----
  load_unit_rowmajor(Ab, lda, M, K, m0 + ar0, ac0, aReg[0]);
  load_unit_rowmajor(Ab, lda, M, K, m0 + ar1, ac1, aReg[1]);
  if (!bTrans) {
    load_unit_rowmajor(Bb, ldb, N, K, n0 + brw, bcu, bReg);
  } else {
    int gn = n0 + brw;
    #pragma unroll
    for (int j = 0; j < 8; ++j) {
      int gk = bcu + j;
      bReg[j] = (gn < N && gk < K) ? Bb[(long long)gk * ldb + gn] : 0.f;
    }
  }
  store_unit(&As[0][ar0 * 32 + ac0], aReg[0]);
  store_unit(&As[0][ar1 * 32 + ac1], aReg[1]);
  store_unit(&Bs[0][brw * 32 + bcu], bReg);
  __syncthreads();

  int cur = 0;
  for (int t = 0; t < nk; ++t) {
    // ---- prefetch chunk t+2 (speculative, near-cache) ----
    if (t + 2 < nk) {
      int pr = m0 + (tid >> 1);
      if (pr < M)
        __builtin_prefetch(&Ab[(long long)pr * lda + (t + 2) * 32 + ((tid & 1) << 4)], 0, 3);
    }
    // ---- fetch chunk t+1 into registers (overlaps WMMA below) ----
    const bool more = (t + 1 < nk);
    if (more) {
      int kc = (t + 1) * 32;
      load_unit_rowmajor(Ab, lda, M, K, m0 + ar0, kc + ac0, aReg[0]);
      load_unit_rowmajor(Ab, lda, M, K, m0 + ar1, kc + ac1, aReg[1]);
      if (!bTrans) {
        load_unit_rowmajor(Bb, ldb, N, K, n0 + brw, kc + bcu, bReg);
      } else {
        int gn = n0 + brw;
        #pragma unroll
        for (int j = 0; j < 8; ++j) {
          int gk = kc + bcu + j;
          bReg[j] = (gn < N && gk < K) ? Bb[(long long)gk * ldb + gn] : 0.f;
        }
      }
    }

    // ---- fragments + 4x WMMA from buffer `cur` ----
    const unsigned short* Ac = As[cur];
    const unsigned short* Bc = Bs[cur];
    Frag16 fa[2], fb[2];
    #pragma unroll
    for (int s = 0; s < 2; ++s) {
      // A 16x32: lane<16 holds row, K {0..7,16..23}; lane>=16 K {8..15,24..31}
      int ao = (wm * 32 + s * 16 + hr) * 32 + hi * 8;
      fa[s].q[0] = *(const u32x4*)(&Ac[ao]);
      fa[s].q[1] = *(const u32x4*)(&Ac[ao + 16]);
      // B 32x16: lane<16 holds col n, K {0..7,16..23}; lane>=16 K {8..15,24..31}
      int bo = (wn * 32 + s * 16 + hr) * 32 + hi * 8;
      fb[s].q[0] = *(const u32x4*)(&Bc[bo]);
      fb[s].q[1] = *(const u32x4*)(&Bc[bo + 16]);
    }
    #pragma unroll
    for (int mi = 0; mi < 2; ++mi)
      #pragma unroll
      for (int ni = 0; ni < 2; ++ni)
        acc[mi][ni] = __builtin_amdgcn_wmma_f32_16x16x32_bf16(
            false, fa[mi].v, false, fb[ni].v, (short)0, acc[mi][ni], false, false);

    // ---- commit chunk t+1 to the alternate buffer ----
    if (more) {
      int nxt = cur ^ 1;
      store_unit(&As[nxt][ar0 * 32 + ac0], aReg[0]);
      store_unit(&As[nxt][ar1 * 32 + ac1], aReg[1]);
      store_unit(&Bs[nxt][brw * 32 + bcu], bReg);
      __syncthreads();
      cur = nxt;
    }
  }

  // epilogue: C/D layout -> VGPR r: M = r + 8*hi, N = lane&15
  #pragma unroll
  for (int mi = 0; mi < 2; ++mi) {
    #pragma unroll
    for (int ni = 0; ni < 2; ++ni) {
      int gn = n0 + wn * 32 + ni * 16 + hr;
      #pragma unroll
      for (int r = 0; r < 8; ++r) {
        int gm = m0 + wm * 32 + mi * 16 + hi * 8 + r;
        if (gm < M && gn < N) {
          float v = acc[mi][ni][r];
          if (bias) v += bias[gn];
          if (act == 1) v = 0.5f * v * (1.0f + erff(v * 0.70710678118654752f)); // exact gelu
          if (residual) v += residual[(long long)gm * ldc + gn];
          Cb[(long long)gm * ldc + gn] = v;
        }
      }
    }
  }
}

// =====================================================================
// LayerNorm over last dim (D), one block (8 waves) per row
// =====================================================================
__global__ __launch_bounds__(256) void layernorm_k(
    const float* __restrict__ x, const float* __restrict__ g,
    const float* __restrict__ b, float* __restrict__ o, int D)
{
  long long row = blockIdx.x;
  const float* xr = x + row * D;
  __shared__ float redS[8], redQ[8];
  __shared__ float s_mu, s_rs;
  float s = 0.f, q = 0.f;
  for (int i = threadIdx.x; i < D; i += 256) { float v = xr[i]; s += v; q += v * v; }
  for (int off = 16; off > 0; off >>= 1) { s += __shfl_xor(s, off); q += __shfl_xor(q, off); }
  int wv = threadIdx.x >> 5;
  if ((threadIdx.x & 31) == 0) { redS[wv] = s; redQ[wv] = q; }
  __syncthreads();
  if (threadIdx.x == 0) {
    float ts = 0.f, tq = 0.f;
    for (int i = 0; i < 8; ++i) { ts += redS[i]; tq += redQ[i]; }
    float mu = ts / D;
    s_mu = mu;
    s_rs = rsqrtf(tq / D - mu * mu + 1e-5f);
  }
  __syncthreads();
  float mu = s_mu, rs = s_rs;
  float* orow = o + row * D;
  for (int i = threadIdx.x; i < D; i += 256)
    orow[i] = (xr[i] - mu) * rs * g[i] + b[i];
}

// =====================================================================
// Causal depthwise conv1d (K=4) + SiLU.  x = first DINNER cols of xz
// [token,4096].  rev!=0 runs on the time-reversed sequence; output xc is
// stored in scan-step order (l' = reversed time for the backward pass).
// =====================================================================
__global__ __launch_bounds__(256) void conv_silu_k(
    const float* __restrict__ xz, const float* __restrict__ w,
    const float* __restrict__ bias, float* __restrict__ xc, int rev)
{
  long long i = (long long)blockIdx.x * 256 + threadIdx.x;
  if (i >= (long long)MTOK * DINNER) return;
  int d = (int)(i % DINNER);
  long long t = i / DINNER;       // token = b*L + l'
  int l = (int)(t % LSEQ);
  long long b = t / LSEQ;
  float acc = bias[d];
  #pragma unroll
  for (int k = 0; k < 4; ++k) {
    int ls = l - 3 + k;            // causal left pad 3
    if (ls >= 0) {
      int lr = rev ? (LSEQ - 1 - ls) : ls;
      acc += w[d * 4 + k] * xz[(b * LSEQ + lr) * (2 * DINNER) + d];
    }
  }
  xc[i] = acc / (1.f + __expf(-acc));    // SiLU
}

// =====================================================================
// Selective scan: one thread per (b,d); 16 states in registers; L=150
// sequential steps.  rev!=0 scatters outputs back to real time order and
// accumulates (fwd pass stores).  Applies Dskip and SiLU(z) gating.
// =====================================================================
__global__ __launch_bounds__(256) void ssm_scan_k(
    const float* __restrict__ xc,   const float* __restrict__ dtb,
    const float* __restrict__ xdbl, const float* __restrict__ A_log,
    const float* __restrict__ Dsk,  const float* __restrict__ xz,
    float* __restrict__ y, int rev, int accumulate)
{
  int gid = blockIdx.x * 256 + threadIdx.x;
  if (gid >= NBATCH * DINNER) return;
  int b = gid >> 11;              // /DINNER
  int d = gid & (DINNER - 1);

  float Ar[NSTATE];
  #pragma unroll
  for (int n = 0; n < NSTATE; ++n) Ar[n] = -__expf(A_log[d * NSTATE + n]);
  float Dv = Dsk[d];
  float h[NSTATE];
  #pragma unroll
  for (int n = 0; n < NSTATE; ++n) h[n] = 0.f;

  const float* dtp = dtb + (long long)b * LSEQ * DINNER + d;
  const float* up  = xc  + (long long)b * LSEQ * DINNER + d;
  const float* bc  = xdbl + (long long)b * LSEQ * 96;

  for (int l = 0; l < LSEQ; ++l) {
    float dtv = dtp[(long long)l * DINNER];
    dtv = (dtv > 20.f) ? dtv : log1pf(__expf(dtv));      // softplus
    float u  = up[(long long)l * DINNER];
    float du = dtv * u;
    const float* xr = bc + l * 96;
    float yv = 0.f;
    #pragma unroll
    for (int n = 0; n < NSTATE; ++n) {
      float Bt = xr[64 + n], Ct = xr[80 + n];
      h[n] = __expf(dtv * Ar[n]) * h[n] + du * Bt;
      yv += h[n] * Ct;
    }
    int lreal = rev ? (LSEQ - 1 - l) : l;
    float zv = xz[((long long)b * LSEQ + lreal) * (2 * DINNER) + DINNER + d];
    float gate = zv / (1.f + __expf(-zv));
    float outv = (yv + Dv * u) * gate;
    float* yp = y + ((long long)b * LSEQ + lreal) * DINNER + d;
    if (accumulate) *yp += outv; else *yp = outv;
  }
}

// =====================================================================
// Causal softmax over scores [B,H,L,L]; scale 1/sqrt(DHEAD)=1/32.
// One wave per row (8 rows per block).
// =====================================================================
__global__ __launch_bounds__(256) void softmax_causal_k(float* __restrict__ sc)
{
  int row  = blockIdx.x * 8 + (threadIdx.x >> 5);
  int lane = threadIdx.x & 31;
  if (row >= NBATCH * NHEADS * LSEQ) return;
  int qi = row % LSEQ;
  float* p = sc + (long long)row * LSEQ;

  float vals[5];
  float mx = -1e30f;
  #pragma unroll
  for (int t = 0; t < 5; ++t) {
    int j = lane + t * 32;
    float v = -1e9f;
    if (j < LSEQ) {
      v = p[j] * 0.03125f;                 // * DHEAD^-0.5
      if (j > qi) v = -1e9f;               // causal mask
    }
    vals[t] = v;
    mx = fmaxf(mx, v);
  }
  for (int off = 16; off > 0; off >>= 1) mx = fmaxf(mx, __shfl_xor(mx, off));
  float sum = 0.f;
  #pragma unroll
  for (int t = 0; t < 5; ++t) {
    int j = lane + t * 32;
    float e = (j < LSEQ) ? __expf(vals[t] - mx) : 0.f;
    vals[t] = e;
    sum += e;
  }
  for (int off = 16; off > 0; off >>= 1) sum += __shfl_xor(sum, off);
  float inv = 1.f / sum;
  #pragma unroll
  for (int t = 0; t < 5; ++t) {
    int j = lane + t * 32;
    if (j < LSEQ) p[j] = vals[t] * inv;
  }
}

// =====================================================================
extern "C" void kernel_launch(void* const* d_in, const int* in_sizes, int n_in,
                              void* d_out, int out_size, void* d_ws, size_t ws_size,
                              hipStream_t stream)
{
  (void)in_sizes; (void)n_in; (void)out_size; (void)ws_size;

  const float* feat      = (const float*)d_in[0];
  const float* mnorm_g   = (const float*)d_in[1];
  const float* mnorm_b   = (const float*)d_in[2];
  const float* in_proj_w = (const float*)d_in[3];
  const float* conv_w[2]    = {(const float*)d_in[4],  (const float*)d_in[11]};
  const float* conv_bias[2] = {(const float*)d_in[5],  (const float*)d_in[12]};
  const float* x_proj_w[2]  = {(const float*)d_in[6],  (const float*)d_in[13]};
  const float* dt_proj_w[2] = {(const float*)d_in[7],  (const float*)d_in[14]};
  const float* dt_proj_b[2] = {(const float*)d_in[8],  (const float*)d_in[15]};
  const float* A_log[2]     = {(const float*)d_in[9],  (const float*)d_in[16]};
  const float* Dskip[2]     = {(const float*)d_in[10], (const float*)d_in[17]};
  const float* out_proj_w = (const float*)d_in[18];
  const float* anorm_g = (const float*)d_in[19];
  const float* anorm_b = (const float*)d_in[20];
  const float* wqkv    = (const float*)d_in[21];
  const float* wo      = (const float*)d_in[22];
  const float* bo      = (const float*)d_in[23];
  const float* fnorm_g = (const float*)d_in[24];
  const float* fnorm_b = (const float*)d_in[25];
  const float* w1      = (const float*)d_in[26];
  const float* b1      = (const float*)d_in[27];
  const float* w2      = (const float*)d_in[28];
  const float* b2      = (const float*)d_in[29];
  float* out = (float*)d_out;
  float* ws  = (float*)d_ws;

  // workspace layout (floats); arena reused across phases
  float* hbuf  = ws;                    // 2400*1024
  float* feat1 = ws + 2457600;          // 2400*1024
  float* arena = ws + 4915200;
  // phase A (mamba)
  float* xz  = arena;                   // 2400*4096
  float* xc  = arena + 9830400;         // 2400*2048
  float* xdb = xc + 4915200;            // 2400*96
  float* dtb = xdb + 230400;            // 2400*2048
  float* ybf = dtb + 4915200;           // 2400*2048
  // phase B (attention) reuses arena
  float* qkv = arena;                   // 2400*12288
  float* att = arena + 29491200;        // 16*4*150*150
  float* ao  = att + 1440000;           // 2400*4096
  // phase C (ffn) reuses arena
  float* ffh = arena;                   // 2400*4096

  auto gemm = [&](const float* A, const float* B, float* C,
                  const float* bias, const float* res,
                  int M, int N, int K, int lda, int ldb, int ldc, int act) {
    dim3 g((N + 63) / 64, (M + 127) / 128, 1);
    wmma_gemm_bf16<<<g, dim3(256), 0, stream>>>(A, B, C, bias, res,
        M, N, K, lda, ldb, ldc, 0LL, 0LL, 0LL, 0LL, 0LL, 0LL, 1, 0, act);
  };

  const int convBlocks = (MTOK * DINNER + 255) / 256;

  // ---------------- Mamba block ----------------
  layernorm_k<<<MTOK, 256, 0, stream>>>(feat, mnorm_g, mnorm_b, hbuf, DMODEL);
  gemm(hbuf, in_proj_w, xz, nullptr, nullptr, MTOK, 2 * DINNER, DMODEL,
       DMODEL, DMODEL, 2 * DINNER, 0);
  for (int dir = 0; dir < 2; ++dir) {
    conv_silu_k<<<convBlocks, 256, 0, stream>>>(xz, conv_w[dir], conv_bias[dir], xc, dir);
    gemm(xc, x_proj_w[dir], xdb, nullptr, nullptr, MTOK, 96, DINNER,
         DINNER, DINNER, 96, 0);
    gemm(xdb, dt_proj_w[dir], dtb, dt_proj_b[dir], nullptr, MTOK, DINNER, DTRANK,
         96, DTRANK, DINNER, 0);
    ssm_scan_k<<<(NBATCH * DINNER) / 256, 256, 0, stream>>>(
        xc, dtb, xdb, A_log[dir], Dskip[dir], xz, ybf, dir, dir);
  }
  gemm(ybf, out_proj_w, feat1, nullptr, feat, MTOK, DMODEL, DINNER,
       DINNER, DINNER, DMODEL, 0);

  // ---------------- Attention ----------------
  layernorm_k<<<MTOK, 256, 0, stream>>>(feat1, anorm_g, anorm_b, hbuf, DMODEL);
  gemm(hbuf, wqkv, qkv, nullptr, nullptr, MTOK, 3 * AINNER, DMODEL,
       DMODEL, DMODEL, 3 * AINNER, 0);
  {
    // scores[b,h] = Q_bh [L,Dh] * K_bh[L,Dh]^T   (batched over 64 (b,h))
    dim3 g((LSEQ + 63) / 64, (LSEQ + 127) / 128, NBATCH * NHEADS);
    wmma_gemm_bf16<<<g, dim3(256), 0, stream>>>(qkv, qkv + AINNER, att,
        nullptr, nullptr, LSEQ, LSEQ, DHEAD, 3 * AINNER, 3 * AINNER, LSEQ,
        (long long)LSEQ * 3 * AINNER, (long long)DHEAD,
        (long long)LSEQ * 3 * AINNER, (long long)DHEAD,
        (long long)NHEADS * LSEQ * LSEQ, (long long)LSEQ * LSEQ,
        NHEADS, 0, 0);
  }
  softmax_causal_k<<<(NBATCH * NHEADS * LSEQ) / 8, 256, 0, stream>>>(att);
  {
    // ao[b,l,h*Dh+dh] = attn_bh [L,L] * V_bh [L,Dh]   (bTrans=1, scattered via ldc)
    dim3 g((DHEAD + 63) / 64, (LSEQ + 127) / 128, NBATCH * NHEADS);
    wmma_gemm_bf16<<<g, dim3(256), 0, stream>>>(att, qkv + 2 * AINNER, ao,
        nullptr, nullptr, LSEQ, DHEAD, LSEQ, LSEQ, 3 * AINNER, AINNER,
        (long long)NHEADS * LSEQ * LSEQ, (long long)LSEQ * LSEQ,
        (long long)LSEQ * 3 * AINNER, (long long)DHEAD,
        (long long)LSEQ * AINNER, (long long)DHEAD,
        NHEADS, 1, 0);
  }
  gemm(ao, wo, out, bo, feat1, MTOK, DMODEL, AINNER, AINNER, AINNER, DMODEL, 0);

  // ---------------- FFN ----------------
  layernorm_k<<<MTOK, 256, 0, stream>>>(out, fnorm_g, fnorm_b, hbuf, DMODEL);
  gemm(hbuf, w1, ffh, b1, nullptr, MTOK, MLPH, DMODEL, DMODEL, DMODEL, MLPH, 1);
  gemm(ffh, w2, out, b2, out, MTOK, DMODEL, MLPH, MLPH, MLPH, DMODEL, 0);
}
